// BehlerParrinello_3659312136806
// MI455X (gfx1250) — matrix operand order repaired and checked by reading, not compiled, not run on hardware
//
#include <hip/hip_runtime.h>
#include <hip/hip_bf16.h>

#define N_ATOMS 131072
#define NUM_GS  128
#define HIDDEN  512
#define N_MOL   1024

typedef __attribute__((ext_vector_type(16))) __bf16        v16bf;
typedef __attribute__((ext_vector_type(8)))  float         v8f;
typedef __attribute__((ext_vector_type(8)))  unsigned int  v8u;
typedef __attribute__((ext_vector_type(4)))  int           v4i;
typedef __attribute__((ext_vector_type(8)))  int           v8i;
typedef __attribute__((ext_vector_type(4)))  unsigned int  v4u;

// ---------------- helpers ----------------
__device__ __forceinline__ unsigned bf16pk2_trunc(float a, float b) {   // fast truncate
    unsigned ua = __builtin_bit_cast(unsigned, a);
    unsigned ub = __builtin_bit_cast(unsigned, b);
    return (ua >> 16) | (ub & 0xffff0000u);
}
__device__ __forceinline__ unsigned short bf16_rne(float a) {           // RNE (prep only)
    unsigned ua = __builtin_bit_cast(unsigned, a);
    return (unsigned short)((ua + 0x7fffu + ((ua >> 16) & 1u)) >> 16);
}

#if __has_builtin(__builtin_amdgcn_tanhf)
__device__ __forceinline__ float fast_tanh(float x) {   // native v_tanh_f32
    return __builtin_amdgcn_tanhf(x);
}
#else
__device__ __forceinline__ float fast_tanh(float x) {   // exp2/rcp fallback
    float e = __builtin_amdgcn_exp2f(x * 2.8853900817779268f);
    return 1.0f - 2.0f * __builtin_amdgcn_rcpf(e + 1.0f);
}
#endif

__device__ __forceinline__ v16bf make_frag(uint4 a, uint4 b) {
    v8u u;
    u[0] = a.x; u[1] = a.y; u[2] = a.z; u[3] = a.w;
    u[4] = b.x; u[5] = b.y; u[6] = b.z; u[7] = b.w;
    return __builtin_bit_cast(v16bf, u);
}

__device__ __forceinline__ unsigned lds_off(void* p) {   // generic -> LDS byte offset
    return (unsigned)(uintptr_t)(__attribute__((address_space(3))) char*)p;
}

// ---- async global->LDS (gfx1250 GLOBAL_LOAD_ASYNC_TO_LDS_B128, ASYNCcnt) ----
#if __has_builtin(__builtin_amdgcn_global_load_async_to_lds_b128)
#define BP_ASYNC 1
#else
#define BP_ASYNC 0
#endif

__device__ __forceinline__ void stage16(char* lds_dst, const uint4* gsrc) {
#if BP_ASYNC
    __builtin_amdgcn_global_load_async_to_lds_b128(
        (__attribute__((address_space(1))) v4i*)gsrc,
        (__attribute__((address_space(3))) v4i*)lds_dst, 0, 0);
#else
    *(uint4*)lds_dst = *gsrc;
#endif
}
__device__ __forceinline__ void stage_fence() {
#if BP_ASYNC
#if __has_builtin(__builtin_amdgcn_s_wait_asynccnt)
    __builtin_amdgcn_s_wait_asynccnt(0);
#else
    asm volatile("s_wait_asynccnt 0" ::: "memory");
#endif
#endif
}

// ---- TDM: TENSOR_LOAD_TO_LDS with LDS padding (TENSORcnt) ----
#if __has_builtin(__builtin_amdgcn_tensor_load_to_lds)
#define BP_TDM 1
#if __has_include(<hip/amd_detail/amd_gfx1250_TDM.h>)
#define BP_TDM6 1                            // therock headers -> 6-arg builtin
#else
#define BP_TDM6 0                            // ROCm 7.2 -> 5-arg builtin
#endif
#else
#define BP_TDM 0
#endif

#if BP_TDM
// 2D tile load: dim1 columns of dim0_dw dwords; LDS gets pad_amount(+1) dwords of
// padding after every 2^(pad_interval+1) dwords  (ISA 8.3/8.4 D# layout).
__device__ __forceinline__ void tdm_load_2d(const void* gsrc, unsigned ldsoff,
                                            unsigned dim0_dw, unsigned dim1,
                                            unsigned pad_i, unsigned pad_a) {
    unsigned long long ga = (unsigned long long)(uintptr_t)gsrc;
    v4u g0;
    g0[0] = 1u;                                              // count=1 (valid), user mode
    g0[1] = ldsoff;                                          // lds_addr
    g0[2] = (unsigned)(ga & 0xffffffffu);                    // global_addr[31:0]
    g0[3] = (unsigned)((ga >> 32) & 0x1ffffffu) | (2u << 30);// global_addr[56:32] | type=2
    v8i g1;
    g1[0] = (int)((2u << 16) | (1u << 20) | (pad_i << 22) | (pad_a << 25)); // 4B, pad on
    g1[1] = (int)((dim0_dw & 0xffffu) << 16);                // tensor_dim0[15:0] @bit48
    g1[2] = (int)((dim0_dw >> 16) | ((dim1 & 0xffffu) << 16)); // dim0[31:16] | dim1[15:0]
    g1[3] = (int)((dim1 >> 16) | ((dim0_dw & 0xffffu) << 16)); // dim1[31:16] | tile_dim0
    g1[4] = (int)(dim1 & 0xffffu);                           // tile_dim1 (tile_dim2 = 0)
    g1[5] = (int)dim0_dw;                                    // tensor_dim0_stride[31:0]
    g1[6] = 0;
    g1[7] = 0;
    v4i z4 = {0, 0, 0, 0};
#if BP_TDM6
    v8i z8 = {0, 0, 0, 0, 0, 0, 0, 0};
    __builtin_amdgcn_tensor_load_to_lds(g0, g1, z4, z4, z8, 0);
#else
    __builtin_amdgcn_tensor_load_to_lds(g0, g1, z4, z4, 0);
#endif
}
__device__ __forceinline__ void tdm_fence() {
#if __has_builtin(__builtin_amdgcn_s_wait_tensorcnt)
    __builtin_amdgcn_s_wait_tensorcnt(0);
#else
    asm volatile("s_wait_tensorcnt 0" ::: "memory");
#endif
}
#endif

// One weight panel -> LDS (TDM if available, else per-lane async copies)
__device__ __forceinline__ void stage_panel(char* dst, const unsigned short* gsrc,
                                            unsigned dim0_dw, unsigned dim1,
                                            unsigned pad_i, unsigned pad_a,
                                            unsigned col_stride_b, int tid) {
#if BP_TDM
    if (tid == 0)
        tdm_load_2d(gsrc, lds_off(dst), dim0_dw, dim1, pad_i, pad_a);
    tdm_fence();
#else
    const uint4* src = (const uint4*)gsrc;
    const int cpc = (int)(dim0_dw / 4);                  // 16B chunks per column
    for (int i = tid; i < (int)dim1 * cpc; i += 128) {
        int col = i / cpc, ch = i % cpc;
        stage16(dst + col * col_stride_b + ch * 16, src + i);
    }
    stage_fence();
#endif
}

// ---------------- LDS layout (bytes) ----------------
// h1 : [2 experts][64 rows][520 bf16]          = 133120 B  (row stride 1040B, 16B aligned)
// wbuf: W1 stage (512 cols x 272B) = 139264 B  OR  W2 panel (128 cols x 1040B) = 133120 B
// b1 : [2][512] f32 = 4096 ; b2 : 4096 ; w3 : 4096
#define H1_ROW_B   1040
#define OFF_H1     0
#define OFF_W      133120
#define OFF_B1     272384
#define OFF_B2     276480
#define OFF_W3     280576
#define SMEM_BYTES 284672

// ---------------- prep: f32 -> bf16, transposed weights ----------------
__global__ void bp_prep(const float* __restrict__ W1A, const float* __restrict__ W1B,
                        const float* __restrict__ W2A, const float* __restrict__ W2B,
                        unsigned short* __restrict__ W1t, unsigned short* __restrict__ W2t) {
    int idx = blockIdx.x * 256 + threadIdx.x;
    if (idx < 2 * HIDDEN * NUM_GS) {                 // W1t[e][col][k] = W1_e[k][col]
        int e = idx >> 16;
        int r = idx & 65535;
        int col = r >> 7, k = r & 127;
        const float* W = e ? W1B : W1A;
        W1t[idx] = bf16_rne(W[k * HIDDEN + col]);
    } else {
        int r = idx - 2 * HIDDEN * NUM_GS;
        if (r < 2 * HIDDEN * HIDDEN) {               // W2t[e][col][k] = W2_e[k][col]
            int e = r >> 18;
            int q = r & 262143;
            int col = q >> 9, k = q & 511;
            const float* W = e ? W2B : W2A;
            W2t[r] = bf16_rne(W[k * HIDDEN + col]);
        }
    }
}

__global__ void bp_zero(float* __restrict__ p, int n) {
    int i = blockIdx.x * 256 + threadIdx.x;
    if (i < n) p[i] = 0.0f;
}

__global__ void bp_finalize(const float* __restrict__ msum, const float* __restrict__ mcnt,
                            float* __restrict__ out) {
    int m = blockIdx.x * 256 + threadIdx.x;
    if (m < N_MOL) out[m] = msum[m] / fmaxf(mcnt[m], 1.0f);
}

// ---------------- fused MoE-MLP kernel ----------------
__global__ __launch_bounds__(128) void bp_mlp(
    const float* __restrict__ Gs, const int* __restrict__ types, const int* __restrict__ mol_id,
    const unsigned short* __restrict__ W1t, const unsigned short* __restrict__ W2t,
    const float* __restrict__ b1A, const float* __restrict__ b1B,
    const float* __restrict__ b2A, const float* __restrict__ b2B,
    const float* __restrict__ w3A, const float* __restrict__ w3B,
    const float* __restrict__ b3A, const float* __restrict__ b3B,
    const float* __restrict__ offA, const float* __restrict__ offB,
    float* __restrict__ mol_sum, float* __restrict__ mol_cnt) {

    extern __shared__ char smem[];
    char*  sh1b = smem + OFF_H1;                     // bf16 h1, row stride 1040B
    char*  swb  = smem + OFF_W;                      // staged weights
    float* sb1  = (float*)(smem + OFF_B1);
    float* sb2  = (float*)(smem + OFF_B2);
    float* sw3  = (float*)(smem + OFF_W3);

    const int tid = threadIdx.x;
    const int w   = tid >> 5;                        // wave id, 0..3 (wave32)
    const int l   = tid & 31;
    const int h   = l >> 4;                          // half-wave select
    const int ln  = l & 15;
    const int a0  = blockIdx.x * 64;

    // stage biases + W3 (f32)
    for (int i = tid; i < HIDDEN; i += 128) {
        sb1[i] = b1A[i]; sb1[HIDDEN + i] = b1B[i];
        sb2[i] = b2A[i]; sb2[HIDDEN + i] = b2B[i];
        sw3[i] = w3A[i]; sw3[HIDDEN + i] = w3B[i];
    }

    // ---- load G A-fragments into VGPRs (16x32 bf16, ISA layout) ----
    // lane: seg0 = K[c*32 + h*8 .. +7], seg1 = K[c*32 + 16 + h*8 .. +7]
    const float* grow = Gs + (size_t)(a0 + w * 16 + ln) * NUM_GS;
    v16bf afrag[4];
#pragma unroll
    for (int c = 0; c < 4; ++c) {
        float4 f0 = *(const float4*)(grow + c * 32 + h * 8);
        float4 f1 = *(const float4*)(grow + c * 32 + h * 8 + 4);
        float4 f2 = *(const float4*)(grow + c * 32 + 16 + h * 8);
        float4 f3 = *(const float4*)(grow + c * 32 + 16 + h * 8 + 4);
        v8u u;
        u[0] = bf16pk2_trunc(f0.x, f0.y); u[1] = bf16pk2_trunc(f0.z, f0.w);
        u[2] = bf16pk2_trunc(f1.x, f1.y); u[3] = bf16pk2_trunc(f1.z, f1.w);
        u[4] = bf16pk2_trunc(f2.x, f2.y); u[5] = bf16pk2_trunc(f2.z, f2.w);
        u[6] = bf16pk2_trunc(f3.x, f3.y); u[7] = bf16pk2_trunc(f3.z, f3.w);
        afrag[c] = __builtin_bit_cast(v16bf, u);
    }

    // ================= layer 1: h1 = tanh(G @ W1_e + b1_e) =================
#pragma unroll
    for (int e = 0; e < 2; ++e) {
        __syncthreads();
        // stage W1t[e] (512 cols x 64 dw) -> LDS, col stride 272B (pad 64dw+4dw)
        stage_panel(swb, W1t + e * 65536, 64, 512, 5, 3, 272, tid);
        __syncthreads();

        for (int j = 0; j < 32; ++j) {               // 32 N-tiles of 16
            float bb = sb1[e * HIDDEN + j * 16 + ln];
            v8f acc;
#pragma unroll
            for (int i = 0; i < 8; ++i) acc[i] = bb;
#pragma unroll
            for (int c = 0; c < 4; ++c) {            // K = 128 in 4 chunks
                const char* bp = swb + (j * 16 + ln) * 272 + (c * 32 + h * 8) * 2;
                v16bf bf = make_frag(*(const uint4*)bp, *(const uint4*)(bp + 32));
                acc = __builtin_amdgcn_wmma_f32_16x16x32_bf16(
                        false, afrag[c], false, bf, (short)0, acc, false, false);
            }
            // tanh + store h1 (truncating bf16 -> plain d16_hi stores)
#pragma unroll
            for (int i = 0; i < 8; ++i) {
                int M = i + h * 8;                   // C/D layout: lanes16-31 -> M+8
                unsigned t = __builtin_bit_cast(unsigned, fast_tanh(acc[i]));
                *(unsigned short*)(sh1b + (size_t)(e * 64 + w * 16 + M) * H1_ROW_B
                                   + (j * 16 + ln) * 2) = (unsigned short)(t >> 16);
            }
        }
    }

    // ====== layer 2 + 3: acc_e += tanh(h1_e @ W2_e + b2_e) * W3_e (fused) ======
    float laccA[8], laccB[8];
#pragma unroll
    for (int i = 0; i < 8; ++i) { laccA[i] = 0.0f; laccB[i] = 0.0f; }

#pragma unroll
    for (int e = 0; e < 2; ++e) {
        for (int nc = 0; nc < 4; ++nc) {             // stream W2 in 128-col panels
            __syncthreads();
            // 128 cols x 256 dw -> LDS, col stride 1040B (pad 256dw+4dw)
            stage_panel(swb, W2t + ((size_t)e * HIDDEN + nc * 128) * 512,
                        256, 128, 7, 3, H1_ROW_B, tid);
            __syncthreads();
            if (nc < 3)                              // hint next panel into L2
                __builtin_prefetch((const char*)(W2t +
                    ((size_t)e * HIDDEN + (nc + 1) * 128) * 512) + tid * 1024, 0, 0);

            // 8 accumulator tiles; K-chunk outer so each A-frag is loaded once
            v8f acc[8];
#pragma unroll
            for (int j = 0; j < 8; ++j) {
                float bb = sb2[e * HIDDEN + nc * 128 + j * 16 + ln];
#pragma unroll
                for (int i = 0; i < 8; ++i) acc[j][i] = bb;
            }
            const char* ap = sh1b + (size_t)(e * 64 + w * 16 + ln) * H1_ROW_B;
            for (int c = 0; c < 16; ++c) {           // K = 512 in 16 chunks
                v16bf af = make_frag(
                    *(const uint4*)(ap + (c * 32 + h * 8) * 2),
                    *(const uint4*)(ap + (c * 32 + 16 + h * 8) * 2));
#pragma unroll
                for (int j = 0; j < 8; ++j) {
                    const char* bp = swb + (j * 16 + ln) * H1_ROW_B + (c * 32 + h * 8) * 2;
                    v16bf bf = make_frag(*(const uint4*)bp, *(const uint4*)(bp + 32));
                    acc[j] = __builtin_amdgcn_wmma_f32_16x16x32_bf16(
                            false, af, false, bf, (short)0, acc[j], false, false);
                }
            }
#pragma unroll
            for (int j = 0; j < 8; ++j) {
                float w3v = sw3[e * HIDDEN + nc * 128 + j * 16 + ln];
#pragma unroll
                for (int i = 0; i < 8; ++i) {
                    float t = fast_tanh(acc[j][i]) * w3v;
                    if (e == 0) laccA[i] += t; else laccB[i] += t;
                }
            }
        }
    }

    // ---- cross-lane N-reduction (16-lane halves), select expert, atomic segment sum ----
    float eb3A = b3A[0] + offA[0];
    float eb3B = b3B[0] + offB[0];
#pragma unroll
    for (int i = 0; i < 8; ++i) {
        float sA = laccA[i], sB = laccB[i];
#pragma unroll
        for (int m = 8; m >= 1; m >>= 1) {
            sA += __shfl_xor(sA, m, 16);
            sB += __shfl_xor(sB, m, 16);
        }
        if (ln == i) {
            int atom = a0 + w * 16 + h * 8 + i;
            float val = (types[atom] == 0) ? (sA + eb3A) : (sB + eb3B);
            int m = mol_id[atom];
            atomicAdd(&mol_sum[m], val);
            atomicAdd(&mol_cnt[m], 1.0f);
        }
    }
}

// ---------------- host entry ----------------
extern "C" void kernel_launch(void* const* d_in, const int* in_sizes, int n_in,
                              void* d_out, int out_size, void* d_ws, size_t ws_size,
                              hipStream_t stream) {
    const float* Gs     = (const float*)d_in[0];
    const int*   types  = (const int*)d_in[1];
    const int*   mol_id = (const int*)d_in[2];
    const float* W1A = (const float*)d_in[3];
    const float* b1A = (const float*)d_in[4];
    const float* W2A = (const float*)d_in[5];
    const float* b2A = (const float*)d_in[6];
    const float* W3A = (const float*)d_in[7];
    const float* b3A = (const float*)d_in[8];
    const float* ofA = (const float*)d_in[9];
    const float* W1B = (const float*)d_in[10];
    const float* b1B = (const float*)d_in[11];
    const float* W2B = (const float*)d_in[12];
    const float* b2B = (const float*)d_in[13];
    const float* W3B = (const float*)d_in[14];
    const float* b3B = (const float*)d_in[15];
    const float* ofB = (const float*)d_in[16];

    float*          mol_sum = (float*)d_ws;                       // 4 KB
    float*          mol_cnt = mol_sum + N_MOL;                    // 4 KB
    unsigned short* W1t = (unsigned short*)((char*)d_ws + 8192);  // 256 KB
    unsigned short* W2t = W1t + 2 * HIDDEN * NUM_GS;              // 1 MB

    bp_zero<<<(2 * N_MOL + 255) / 256, 256, 0, stream>>>(mol_sum, 2 * N_MOL);

    int prep_elems = 2 * HIDDEN * NUM_GS + 2 * HIDDEN * HIDDEN;   // 655360
    bp_prep<<<(prep_elems + 255) / 256, 256, 0, stream>>>(W1A, W1B, W2A, W2B, W1t, W2t);

    bp_mlp<<<N_ATOMS / 64, 128, SMEM_BYTES, stream>>>(
        Gs, types, mol_id, W1t, W2t,
        b1A, b1B, b2A, b2B, W3A, W3B, b3A, b3B, ofA, ofB,
        mol_sum, mol_cnt);

    bp_finalize<<<(N_MOL + 255) / 256, 256, 0, stream>>>(mol_sum, mol_cnt, (float*)d_out);
}
